// GPT_51814485458988
// MI455X (gfx1250) — compile-verified, matrix operand order checked by reading, not run on hardware
//
#include <hip/hip_runtime.h>

// ---------------- model constants ----------------
#define LNUM   12
#define HNUM   12
#define DMODEL 768
#define VOCAB  50257
#define BATCH  4
#define SEQ    1024
#define MROWS  (BATCH * SEQ)   // 4096
#define D3     (3 * DMODEL)    // 2304
#define FDIM   (4 * DMODEL)    // 3072
#define HDIM   64
#define EPSLN  1e-5f
#define BSTRIDE 40             // padded LDS row stride (halves), 80B = 16B-aligned

typedef _Float16 half_t;
typedef __attribute__((ext_vector_type(16))) _Float16 v16h;
typedef __attribute__((ext_vector_type(8)))  _Float16 v8h;
typedef __attribute__((ext_vector_type(8)))  float    v8f;
typedef __attribute__((ext_vector_type(4)))  float    v4f;

// ---------------- gfx1250 async copy (ASYNCcnt path) ----------------
__device__ __forceinline__ void async_copy_b128(void* lds_dst, const void* gsrc) {
  const unsigned lds_off = (unsigned)(unsigned long long)lds_dst;  // low 32b = LDS offset
  asm volatile("global_load_async_to_lds_b128 %0, %1, off"
               :: "v"(lds_off), "v"(gsrc) : "memory");
}
__device__ __forceinline__ void wait_async() {
  asm volatile("s_wait_asynccnt 0x0" ::: "memory");
}

// ---------------- WMMA fragment loaders ----------------
// 16-bit A-matrix 16x32 layout (ISA 7.12.2): lane<16 holds K = {0..7,16..23},
// lane>=16 holds K = {8..15,24..31}; halves packed in VGPR order. The same
// per-lane index pattern serves a B-fragment when rowbase points at the
// contiguous K-run of W^T's row n.
__device__ __forceinline__ v16h load_frag_rowmajor(const float* rowbase, int k0, float scale) {
  const int lane = threadIdx.x & 31;
  const int kb = k0 + ((lane & 16) ? 8 : 0);
  const float* p = rowbase + kb;
  const v4f f0 = *(const v4f*)(p + 0);
  const v4f f1 = *(const v4f*)(p + 4);
  const v4f f2 = *(const v4f*)(p + 16);
  const v4f f3 = *(const v4f*)(p + 20);
  v16h a;
#pragma unroll
  for (int j = 0; j < 4; ++j) {
    a[j + 0]  = (half_t)(f0[j] * scale);
    a[j + 4]  = (half_t)(f1[j] * scale);
    a[j + 8]  = (half_t)(f2[j] * scale);
    a[j + 12] = (half_t)(f3[j] * scale);
  }
  return a;
}

// B-fragment from LDS tile stored transposed as f16: Bs[n][k], row stride BSTRIDE.
// Two 16-byte ds_load_b128 per fragment.
__device__ __forceinline__ v16h load_bfrag_lds(const half_t* Bs, int nrow0) {
  const int lane = threadIdx.x & 31;
  const int kb = (lane & 16) ? 8 : 0;
  const half_t* p = Bs + (nrow0 + (lane & 15)) * BSTRIDE + kb;
  const v8h lo = *(const v8h*)(p);
  const v8h hi = *(const v8h*)(p + 16);
  v16h b;
#pragma unroll
  for (int j = 0; j < 8; ++j) { b[j] = lo[j]; b[j + 8] = hi[j]; }
  return b;
}

__device__ __forceinline__ float gelu_tanh(float v) {
  const float t = 0.7978845608028654f * (v + 0.044715f * v * v * v);
  return 0.5f * v * (1.0f + tanhf(t));
}

// ---------------- NN GEMM: C = act(A @ W + bias) (+ res),  W is [K,N] ----------
// Block: 128 threads (4 waves). Block tile: 16(M) x 256(N); each wave owns a
// 16x64 strip = 4 accumulators sharing one A fragment (4 WMMAs per k-step).
// A tile (16x32 f32) staged via async-copy to LDS; B tile (32x256) staged
// coalesced and transposed to f16 in LDS.
__global__ __launch_bounds__(128) void gemm_nn(const float* __restrict__ A,
                                               const float* __restrict__ W,
                                               const float* __restrict__ bias,
                                               const float* __restrict__ res,
                                               float* __restrict__ C,
                                               int N, int K, int act) {
  __shared__ float  As[16 * 32];          // 2 KB, f32
  __shared__ half_t Bs[256 * BSTRIDE];    // 20 KB, f16 transposed [n][k]
  const int lane = threadIdx.x & 31;
  const int wave = threadIdx.x >> 5;
  const int m0 = blockIdx.x * 16;
  const int nblk = blockIdx.y * 256;
  const bool full = (nblk + 256 <= N);

  const int tr = threadIdx.x >> 3;        // A-stage row 0..15
  const int tc = (threadIdx.x & 7) * 4;   // A-stage k-chunk

  v8f acc[4];
#pragma unroll
  for (int t = 0; t < 4; ++t) acc[t] = (v8f){0.f, 0.f, 0.f, 0.f, 0.f, 0.f, 0.f, 0.f};

  for (int k0 = 0; k0 < K; k0 += 32) {
    // --- stage A tile: one async b128 per thread (memory -> LDS, no VGPR data)
    async_copy_b128(&As[tr * 32 + tc], A + (size_t)(m0 + tr) * K + (k0 + tc));

    // --- stage B tile 32x256, coalesced b128 reads, transpose to f16 in LDS
    if (full) {
#pragma unroll
      for (int it = 0; it < 16; ++it) {
        const int cidx = it * 128 + threadIdx.x;  // 0..2047 float4-chunks
        const int k = cidx >> 6;                  // 0..31
        const int n4 = (cidx & 63) * 4;           // 0..252
        const v4f f = *(const v4f*)(W + (size_t)(k0 + k) * N + nblk + n4);
#pragma unroll
        for (int j = 0; j < 4; ++j) Bs[(n4 + j) * BSTRIDE + k] = (half_t)f[j];
      }
    } else {
      // tail block: clamp column index (garbage only lands in discarded columns)
#pragma unroll
      for (int it = 0; it < 16; ++it) {
        const int cidx = it * 128 + threadIdx.x;
        const int k = cidx >> 6;
        const int n4 = (cidx & 63) * 4;
#pragma unroll
        for (int j = 0; j < 4; ++j) {
          int n = nblk + n4 + j;
          n = n < N ? n : N - 1;
          Bs[(n4 + j) * BSTRIDE + k] = (half_t)W[(size_t)(k0 + k) * N + n];
        }
      }
    }
    wait_async();
    __syncthreads();

    const v16h af = load_frag_rowmajor(&As[(lane & 15) * 32], 0, 1.0f);
#pragma unroll
    for (int t = 0; t < 4; ++t) {
      const v16h bf = load_bfrag_lds(Bs, wave * 64 + t * 16);
      acc[t] = __builtin_amdgcn_wmma_f32_16x16x32_f16(false, af, false, bf, (short)0,
                                                      acc[t], false, false);
    }
    __syncthreads();
  }

  // --- fused epilogue
  const int mb = m0 + ((lane & 16) ? 8 : 0);
#pragma unroll
  for (int t = 0; t < 4; ++t) {
    const int n = nblk + wave * 64 + t * 16 + (lane & 15);
    if (n < N) {
      const float bv = bias ? bias[n] : 0.0f;
#pragma unroll
      for (int i = 0; i < 8; ++i) {
        const size_t oidx = (size_t)(mb + i) * N + n;
        float v = acc[t][i] + bv;
        if (act == 1) v = gelu_tanh(v);
        if (res) v += res[oidx];
        C[oidx] = v;
      }
    }
  }
}

// ---------------- BT GEMM (tied lm_head): C = A @ Wt^T, Wt is [N,K] -----------
// Contiguous per-lane K runs -> direct global b128 loads; 16x64 strip per wave.
__global__ __launch_bounds__(128) void gemm_bt(const float* __restrict__ A,
                                               const float* __restrict__ Wt,
                                               float* __restrict__ C,
                                               int N, int K) {
  const int lane = threadIdx.x & 31;
  const int wave = threadIdx.x >> 5;
  const int m0 = blockIdx.x * 16;
  const int n0 = blockIdx.y * 256 + wave * 64;
  if (n0 >= N) return;  // uniform per wave: EXEC all-ones for WMMA

  const float* Arow = A + (size_t)(m0 + (lane & 15)) * K;
  const float* wrow[4];
  int ncol[4];
#pragma unroll
  for (int t = 0; t < 4; ++t) {
    const int n = n0 + t * 16 + (lane & 15);
    ncol[t] = n;
    const int ncl = n < N ? n : N - 1;  // clamp addr; stores are guarded
    wrow[t] = Wt + (size_t)ncl * K;
  }

  v8f acc[4];
#pragma unroll
  for (int t = 0; t < 4; ++t) acc[t] = (v8f){0.f, 0.f, 0.f, 0.f, 0.f, 0.f, 0.f, 0.f};

  for (int k0 = 0; k0 < K; k0 += 32) {
    const v16h af = load_frag_rowmajor(Arow, k0, 1.0f);
#pragma unroll
    for (int t = 0; t < 4; ++t) {
      const v16h bf = load_frag_rowmajor(wrow[t], k0, 1.0f);
      acc[t] = __builtin_amdgcn_wmma_f32_16x16x32_f16(false, af, false, bf, (short)0,
                                                      acc[t], false, false);
    }
  }

  const int mb = m0 + ((lane & 16) ? 8 : 0);
#pragma unroll
  for (int t = 0; t < 4; ++t) {
    if (ncol[t] < N) {
#pragma unroll
      for (int i = 0; i < 8; ++i)
        C[(size_t)(mb + i) * N + ncol[t]] = acc[t][i];
    }
  }
}

// ---------------- flash-style causal attention (one wave per b,h,16-q tile) ---
__global__ __launch_bounds__(32) void attn_wmma(const float* __restrict__ qkv,
                                                float* __restrict__ y) {
  __shared__ _Float16 pst[16 * 32];  // staged P tile (16 q-rows x 32 keys)
  const int lane = threadIdx.x & 31;
  const int qt = blockIdx.x % (SEQ / 16);
  const int bh = blockIdx.x / (SEQ / 16);
  const int h = bh % HNUM;
  const int b = bh / HNUM;
  const int q0 = qt * 16;
  const size_t rs = (size_t)D3;
  const float* base = qkv + (size_t)b * SEQ * rs;
  const int hi = (lane & 16) ? 8 : 0;

  const int qrow = q0 + (lane & 15);
  const float* qp = base + (size_t)qrow * rs + h * HDIM;
  const v16h aq0 = load_frag_rowmajor(qp, 0, 0.125f);   // hd^-0.5 folded in
  const v16h aq1 = load_frag_rowmajor(qp, 32, 0.125f);

  float mrow[8], lrow[8];
  v8f o[4];
#pragma unroll
  for (int dc = 0; dc < 4; ++dc) o[dc] = (v8f){0.f, 0.f, 0.f, 0.f, 0.f, 0.f, 0.f, 0.f};
#pragma unroll
  for (int i = 0; i < 8; ++i) { mrow[i] = -3.0e38f; lrow[i] = 0.0f; }

  for (int kt = 0; kt < q0 + 16; kt += 32) {
    v8f s0 = {0.f, 0.f, 0.f, 0.f, 0.f, 0.f, 0.f, 0.f};
    v8f s1 = {0.f, 0.f, 0.f, 0.f, 0.f, 0.f, 0.f, 0.f};
    {
      const int t0 = kt + (lane & 15);
      const int tc0 = t0 < SEQ ? t0 : SEQ - 1;
      const float* kp0 = base + (size_t)tc0 * rs + DMODEL + h * HDIM;
      s0 = __builtin_amdgcn_wmma_f32_16x16x32_f16(false, aq0, false,
             load_frag_rowmajor(kp0, 0, 1.0f), (short)0, s0, false, false);
      s0 = __builtin_amdgcn_wmma_f32_16x16x32_f16(false, aq1, false,
             load_frag_rowmajor(kp0, 32, 1.0f), (short)0, s0, false, false);
      const int t1 = t0 + 16;
      const int tc1 = t1 < SEQ ? t1 : SEQ - 1;
      const float* kp1 = base + (size_t)tc1 * rs + DMODEL + h * HDIM;
      s1 = __builtin_amdgcn_wmma_f32_16x16x32_f16(false, aq0, false,
             load_frag_rowmajor(kp1, 0, 1.0f), (short)0, s1, false, false);
      s1 = __builtin_amdgcn_wmma_f32_16x16x32_f16(false, aq1, false,
             load_frag_rowmajor(kp1, 32, 1.0f), (short)0, s1, false, false);
    }

    const int c0 = kt + (lane & 15);
#pragma unroll
    for (int i = 0; i < 8; ++i) {
      const int q = q0 + hi + i;
      float v0 = (c0 <= q) ? s0[i] : -3.0e38f;
      float v1 = (c0 + 16 <= q) ? s1[i] : -3.0e38f;
      float mx = fmaxf(v0, v1);
#pragma unroll
      for (int msk = 1; msk < 16; msk <<= 1) mx = fmaxf(mx, __shfl_xor(mx, msk));
      const float nm = fmaxf(mrow[i], mx);
      const float alpha = __expf(mrow[i] - nm);
      const float p0 = __expf(v0 - nm);
      const float p1 = __expf(v1 - nm);
      float ps = p0 + p1;
#pragma unroll
      for (int msk = 1; msk < 16; msk <<= 1) ps += __shfl_xor(ps, msk);
      lrow[i] = lrow[i] * alpha + ps;
      mrow[i] = nm;
      o[0][i] *= alpha; o[1][i] *= alpha; o[2][i] *= alpha; o[3][i] *= alpha;
      pst[(hi + i) * 32 + (lane & 15)] = (half_t)p0;
      pst[(hi + i) * 32 + 16 + (lane & 15)] = (half_t)p1;
    }
    __syncthreads();

    const int pkb = (lane & 16) ? 8 : 0;
    const _Float16* pp = &pst[(lane & 15) * 32 + pkb];
    v16h ap;
#pragma unroll
    for (int j = 0; j < 8; ++j) { ap[j] = pp[j]; ap[j + 8] = pp[j + 16]; }

    const float* vbase = base + 2 * DMODEL + h * HDIM;
#pragma unroll
    for (int dc = 0; dc < 4; ++dc) {
      const int d = dc * 16 + (lane & 15);
      v16h bv;
#pragma unroll
      for (int j = 0; j < 8; ++j) {
        int ka = kt + pkb + j;
        int kb2 = kt + pkb + 16 + j;
        ka = ka < SEQ ? ka : SEQ - 1;   // P==0 for OOB/masked keys
        kb2 = kb2 < SEQ ? kb2 : SEQ - 1;
        bv[j]     = (half_t)vbase[(size_t)ka * rs + d];
        bv[j + 8] = (half_t)vbase[(size_t)kb2 * rs + d];
      }
      o[dc] = __builtin_amdgcn_wmma_f32_16x16x32_f16(false, ap, false, bv, (short)0,
                                                     o[dc], false, false);
    }
    __syncthreads();
  }

#pragma unroll
  for (int dc = 0; dc < 4; ++dc) {
    const int d = h * HDIM + dc * 16 + (lane & 15);
#pragma unroll
    for (int i = 0; i < 8; ++i) {
      const int q = q0 + hi + i;
      y[(size_t)(b * SEQ + q) * DMODEL + d] = o[dc][i] / lrow[i];
    }
  }
}

// ---------------- block reductions ----------------
__device__ __forceinline__ float block_sum(float v, float* red) {
  const int tid = threadIdx.x;
  red[tid] = v;
  __syncthreads();
#pragma unroll
  for (int off = 128; off > 0; off >>= 1) {
    if (tid < off) red[tid] += red[tid + off];
    __syncthreads();
  }
  const float r = red[0];
  __syncthreads();
  return r;
}

__device__ __forceinline__ float block_max(float v, float* red) {
  const int tid = threadIdx.x;
  red[tid] = v;
  __syncthreads();
#pragma unroll
  for (int off = 128; off > 0; off >>= 1) {
    if (tid < off) red[tid] = fmaxf(red[tid], red[tid + off]);
    __syncthreads();
  }
  const float r = red[0];
  __syncthreads();
  return r;
}

// ---------------- layernorm ----------------
__global__ __launch_bounds__(256) void ln_kernel(const float* __restrict__ x,
                                                 const float* __restrict__ g,
                                                 const float* __restrict__ bsh,
                                                 float* __restrict__ out) {
  __shared__ float red[256];
  const int row = blockIdx.x;
  const float* xr = x + (size_t)row * DMODEL;
  float s = 0.f;
  for (int i = threadIdx.x; i < DMODEL; i += 256) s += xr[i];
  const float mu = block_sum(s, red) * (1.0f / DMODEL);
  float s2 = 0.f;
  for (int i = threadIdx.x; i < DMODEL; i += 256) {
    const float d = xr[i] - mu;
    s2 += d * d;
  }
  const float rstd = rsqrtf(block_sum(s2, red) * (1.0f / DMODEL) + EPSLN);
  for (int i = threadIdx.x; i < DMODEL; i += 256)
    out[(size_t)row * DMODEL + i] = (xr[i] - mu) * rstd * g[i] + bsh[i];
}

// ---------------- token + position embedding ----------------
__global__ __launch_bounds__(256) void embed_kernel(const int* __restrict__ idx,
                                                    const float* __restrict__ wte,
                                                    const float* __restrict__ wpe,
                                                    float* __restrict__ X) {
  const int i = blockIdx.x * 256 + threadIdx.x;
  if (i >= MROWS * DMODEL) return;
  const int bt = i / DMODEL;
  const int d = i - bt * DMODEL;
  const int t = bt & (SEQ - 1);
  X[i] = wte[(size_t)idx[bt] * DMODEL + d] + wpe[(size_t)t * DMODEL + d];
}

// ---------------- per-row NLL (log-softmax gather), deterministic ----------------
__global__ __launch_bounds__(256) void nll_kernel(const float* __restrict__ logits,
                                                  const int* __restrict__ targets,
                                                  float* __restrict__ nll) {
  __shared__ float red[256];
  const int row = blockIdx.x;
  const float* lr = logits + (size_t)row * VOCAB;
  float mx = -3.0e38f;
  for (int i = threadIdx.x; i < VOCAB; i += 256) mx = fmaxf(mx, lr[i]);
  mx = block_max(mx, red);
  float s = 0.f;
  for (int i = threadIdx.x; i < VOCAB; i += 256) s += __expf(lr[i] - mx);
  s = block_sum(s, red);
  if (threadIdx.x == 0) nll[row] = (logf(s) + mx) - lr[targets[row]];
}

__global__ __launch_bounds__(256) void loss_kernel(const float* __restrict__ nll,
                                                   float* __restrict__ out) {
  __shared__ float red[256];
  float s = 0.f;
  for (int i = threadIdx.x; i < MROWS; i += 256) s += nll[i];
  s = block_sum(s, red);
  if (threadIdx.x == 0) out[0] = s * (1.0f / MROWS);
}

// ---------------- launcher ----------------
extern "C" void kernel_launch(void* const* d_in, const int* in_sizes, int n_in,
                              void* d_out, int out_size, void* d_ws, size_t ws_size,
                              hipStream_t stream) {
  (void)in_sizes; (void)n_in; (void)out_size; (void)ws_size;

  const int* idx     = (const int*)d_in[0];
  const int* targets = (const int*)d_in[1];
  const float* wte   = (const float*)d_in[2];
  const float* wpe   = (const float*)d_in[3];
  const float* ln1_g = (const float*)d_in[4];
  const float* ln1_b = (const float*)d_in[5];
  const float* qkv_w = (const float*)d_in[6];
  const float* qkv_b = (const float*)d_in[7];
  const float* wo_w  = (const float*)d_in[8];
  const float* wo_b  = (const float*)d_in[9];
  const float* ln2_g = (const float*)d_in[10];
  const float* ln2_b = (const float*)d_in[11];
  const float* fc1_w = (const float*)d_in[12];
  const float* fc1_b = (const float*)d_in[13];
  const float* fc2_w = (const float*)d_in[14];
  const float* fc2_b = (const float*)d_in[15];
  const float* lnf_g = (const float*)d_in[16];
  const float* lnf_b = (const float*)d_in[17];
  float* out = (float*)d_out;

  float* ws = (float*)d_ws;
  size_t off = 0;
  float* X   = ws + off; off += (size_t)MROWS * DMODEL;  // residual stream
  float* Hb  = ws + off; off += (size_t)MROWS * DMODEL;  // LN output
  float* QKV = ws + off; off += (size_t)MROWS * D3;      // fused qkv
  float* Y   = ws + off; off += (size_t)MROWS * DMODEL;  // attention output
  float* F1  = ws + off; off += (size_t)MROWS * FDIM;    // MLP hidden
  float* NLL = ws + off; off += (size_t)MROWS;           // per-row nll

  embed_kernel<<<(MROWS * DMODEL + 255) / 256, 256, 0, stream>>>(idx, wte, wpe, X);

  for (int l = 0; l < LNUM; ++l) {
    ln_kernel<<<MROWS, 256, 0, stream>>>(X, ln1_g + l * DMODEL, ln1_b + l * DMODEL, Hb);
    gemm_nn<<<dim3(MROWS / 16, (D3 + 255) / 256), 128, 0, stream>>>(
        Hb, qkv_w + (size_t)l * DMODEL * D3, qkv_b + (size_t)l * D3, nullptr, QKV,
        D3, DMODEL, 0);
    attn_wmma<<<BATCH * HNUM * (SEQ / 16), 32, 0, stream>>>(QKV, Y);
    gemm_nn<<<dim3(MROWS / 16, (DMODEL + 255) / 256), 128, 0, stream>>>(
        Y, wo_w + (size_t)l * DMODEL * DMODEL, wo_b + (size_t)l * DMODEL, X, X,
        DMODEL, DMODEL, 0);
    ln_kernel<<<MROWS, 256, 0, stream>>>(X, ln2_g + l * DMODEL, ln2_b + l * DMODEL, Hb);
    gemm_nn<<<dim3(MROWS / 16, (FDIM + 255) / 256), 128, 0, stream>>>(
        Hb, fc1_w + (size_t)l * DMODEL * FDIM, fc1_b + (size_t)l * FDIM, nullptr, F1,
        FDIM, DMODEL, 1);
    gemm_nn<<<dim3(MROWS / 16, (DMODEL + 255) / 256), 128, 0, stream>>>(
        F1, fc2_w + (size_t)l * FDIM * DMODEL, fc2_b + (size_t)l * DMODEL, X, X,
        DMODEL, FDIM, 0);
  }

  ln_kernel<<<MROWS, 256, 0, stream>>>(X, lnf_g, lnf_b, Hb);
  gemm_bt<<<dim3(MROWS / 16, (VOCAB + 255) / 256), 128, 0, stream>>>(
      Hb, wte, out, VOCAB, DMODEL);

  nll_kernel<<<MROWS, 256, 0, stream>>>(out, targets, NLL);
  loss_kernel<<<1, 256, 0, stream>>>(NLL, out + (size_t)MROWS * VOCAB);
}
